// Network_47227460387322
// MI455X (gfx1250) — compile-verified, matrix operand order checked by reading, not run on hardware
//
#include <hip/hip_runtime.h>
#include <hip/hip_bf16.h>

// ---------------------------------------------------------------------------
// MI455X (gfx1250) implementation.
//
// Roofline: big GEMMs are 12.8 GFLOP over 400 MB streamed -> AI = 32 FLOP/B.
// bf16 WMMA (V_WMMA_F32_16X16X32_BF16) makes them HBM-bound (~17us @23.3TB/s).
// To actually reach that BW we need ~16MB of loads in flight, so the first
// GEMM layer is split 8-ways along K (7504 waves), ping-pong software
// pipelined (no register rotation copies), scalar (SGPR) loop control via
// readfirstlane, global_prefetch on the A stream, f32-atomic partial
// accumulation into an L2-resident h buffer. Graph scatter = L2-resident
// float atomics (emb table 960KB << 192MB L2). Final MLP = one K=32 bf16
// WMMA per 16-row tile.
// ---------------------------------------------------------------------------

typedef __attribute__((ext_vector_type(16))) __bf16 v16bf;
typedef __attribute__((ext_vector_type(8)))  float  v8f;

#define N_USERS 10000
#define M_ITEMS 5000
#define N_NODES (N_USERS + M_ITEMS)
#define EMB 16
#define HID 64
#define KSPLIT 8

// Wave-uniform value -> SGPR (scalar loop control / addressing)
__device__ __forceinline__ int uni(int v) { return __builtin_amdgcn_readfirstlane(v); }

// K-index inside a 32-wide bf16 WMMA K-tile for fragment slot j of a lane.
// A-frag (16x32): lanes 0-15 hold K {0..7,16..23}, lanes 16-31 hold {8..15,24..31}.
// B-frag (32x16) mirrors this with N = lane%16.
__device__ __forceinline__ int kmap(int j, int half) {
    return (j & 7) + ((j & 8) << 1) + half * 8;   // (j%8) + (j>=8?16:0) + half*8
}

__device__ __forceinline__ v16bf loadA_full(const float* __restrict__ xr, int k0, int half) {
    const float4* p = (const float4*)(xr + k0 + half * 8);
    const float4* q = (const float4*)(xr + k0 + 16 + half * 8);
    float4 f0 = p[0], f1 = p[1], f2 = q[0], f3 = q[1];
    v16bf a;
    a[0]=(__bf16)f0.x; a[1]=(__bf16)f0.y; a[2]=(__bf16)f0.z; a[3]=(__bf16)f0.w;
    a[4]=(__bf16)f1.x; a[5]=(__bf16)f1.y; a[6]=(__bf16)f1.z; a[7]=(__bf16)f1.w;
    a[8]=(__bf16)f2.x; a[9]=(__bf16)f2.y; a[10]=(__bf16)f2.z; a[11]=(__bf16)f2.w;
    a[12]=(__bf16)f3.x; a[13]=(__bf16)f3.y; a[14]=(__bf16)f3.z; a[15]=(__bf16)f3.w;
    return a;
}

__device__ __forceinline__ void loadB4(const __bf16* __restrict__ wp, int kt, int lane, v16bf b[4]) {
#pragma unroll
    for (int nt = 0; nt < 4; ++nt)
        b[nt] = *(const v16bf*)(wp + (((size_t)kt * 4 + nt) * 32 + lane) * 16);
}

// Pack a row-major f32 weight W[K][N] into per-lane-contiguous bf16 B-fragments:
// out[((kt*NT + nt)*32 + lane)*16 + j]. Zero-pads K tail.
__global__ void pack_w_kernel(const float* __restrict__ W, int K, int N,
                              int NT, int KT, __bf16* __restrict__ out) {
    int tid = blockIdx.x * blockDim.x + threadIdx.x;
    int total = KT * NT * 32;
    if (tid >= total) return;
    int lane = tid & 31;
    int nt   = (tid >> 5) % NT;
    int kt   = tid / (32 * NT);
    int half = lane >> 4, lm = lane & 15;
    int n = nt * 16 + lm;
    __bf16* o = out + (size_t)tid * 16;
#pragma unroll
    for (int j = 0; j < 16; ++j) {
        int k = kt * 32 + kmap(j, half);
        float v = (k < K) ? W[(size_t)k * N + n] : 0.0f;
        o[j] = (__bf16)v;
    }
}

// First MLP layer, K-split: each wave handles one 16-row M-tile x one K-slice,
// ping-pong software-pipelined A/B fragment loads with SGPR loop control,
// partial sums atomically added into hOut[Mtiles*16][64] (zeroed, L2-resident).
__global__ __launch_bounds__(256)
void gemm1_partial_kernel(const float* __restrict__ X, int M, int K, int KT,
                          const __bf16* __restrict__ w1p, float* __restrict__ hOut) {
    int lane = threadIdx.x & 31;
    int half = lane >> 4, lm = lane & 15;
    int gw = uni(blockIdx.x * 8 + (int)(threadIdx.x >> 5));
    int Mtiles = (M + 15) >> 4;
    if (gw >= Mtiles * KSPLIT) return;
    int tile = gw / KSPLIT;
    int s    = gw % KSPLIT;
    int kper = (KT + KSPLIT - 1) / KSPLIT;
    int kt0 = s * kper;
    int kt1 = kt0 + kper; if (kt1 > KT) kt1 = KT;
    if (kt0 >= kt1) return;
    int row = tile * 16 + lm; if (row >= M) row = M - 1;
    const float* xr = X + (size_t)row * K;

    int KTfull    = K >> 5;                       // tiles with all 32 K valid
    int ktFullEnd = kt1 < KTfull ? kt1 : KTfull;
    int nF = ktFullEnd - kt0;

    v8f acc[4] = {v8f{}, v8f{}, v8f{}, v8f{}};
    v16bf a0 = {}, a1 = {};
    v16bf b0[4] = {}, b1[4] = {};

    int i = 0;
    if (nF > 0) { a0 = loadA_full(xr, kt0 * 32, half); loadB4(w1p, kt0, lane, b0); }
    for (; i + 2 <= nF; i += 2) {                 // ping-pong: no fragment copies
        // iter i: compute buf0, load buf1 for iter i+1 (always exists here)
        a1 = loadA_full(xr, (kt0 + i + 1) * 32, half);
        loadB4(w1p, kt0 + i + 1, lane, b1);
        __builtin_prefetch(xr + (size_t)(kt0 + i + 2) * 32 + half * 8, 0, 1);
#pragma unroll
        for (int nt = 0; nt < 4; ++nt)
            acc[nt] = __builtin_amdgcn_wmma_f32_16x16x32_bf16(
                false, a0, false, b0[nt], (short)0, acc[nt], false, false);
        // iter i+1: compute buf1, load buf0 for iter i+2 (if any)
        if (i + 2 < nF) {
            a0 = loadA_full(xr, (kt0 + i + 2) * 32, half);
            loadB4(w1p, kt0 + i + 2, lane, b0);
            __builtin_prefetch(xr + (size_t)(kt0 + i + 3) * 32 + half * 8, 0, 1);
        }
#pragma unroll
        for (int nt = 0; nt < 4; ++nt)
            acc[nt] = __builtin_amdgcn_wmma_f32_16x16x32_bf16(
                false, a1, false, b1[nt], (short)0, acc[nt], false, false);
    }
    if (i < nF) {                                 // odd remainder (buf0 loaded)
#pragma unroll
        for (int nt = 0; nt < 4; ++nt)
            acc[nt] = __builtin_amdgcn_wmma_f32_16x16x32_bf16(
                false, a0, false, b0[nt], (short)0, acc[nt], false, false);
    }
    if (kt1 > KTfull) {                           // single guarded K-tail tile
        int kt = KTfull;
        v16bf a = {};
#pragma unroll
        for (int j = 0; j < 16; ++j) {
            int k = kt * 32 + kmap(j, half);
            a[j] = (__bf16)((k < K) ? xr[k] : 0.0f);
        }
        v16bf b[4]; loadB4(w1p, kt, lane, b);
#pragma unroll
        for (int nt = 0; nt < 4; ++nt)
            acc[nt] = __builtin_amdgcn_wmma_f32_16x16x32_bf16(
                false, a, false, b[nt], (short)0, acc[nt], false, false);
    }
    // C layout: M = r + 8*half, N = nt*16 + lm. Accumulate partials into h.
#pragma unroll
    for (int nt = 0; nt < 4; ++nt) {
#pragma unroll
        for (int r = 0; r < 8; ++r)
            atomicAdd(&hOut[(size_t)(tile * 16 + r + half * 8) * HID + nt * 16 + lm],
                      acc[nt][r]);
    }
}

// Second MLP layer: emb = relu(relu(h + b1) @ W2 + b2). A-fragments gathered
// straight from global h (row stride 256B -> aligned float4 loads), no LDS.
__global__ __launch_bounds__(256)
void mlp_stage2_kernel(const float* __restrict__ h, int M,
                       const float* __restrict__ b1,
                       const __bf16* __restrict__ w2p, const float* __restrict__ b2,
                       float* __restrict__ embOut) {
    int lane = threadIdx.x & 31;
    int half = lane >> 4, lm = lane & 15;
    int tile = uni(blockIdx.x * 8 + (int)(threadIdx.x >> 5));
    int Mtiles = (M + 15) >> 4;
    if (tile >= Mtiles) return;
    const float* hr = h + (size_t)(tile * 16 + lm) * HID;   // A-frag row = lm
    v8f c2 = v8f{};
#pragma unroll
    for (int kk = 0; kk < 2; ++kk) {
        int kb = kk * 32 + half * 8;
        const float4* p  = (const float4*)(hr + kb);
        const float4* q  = (const float4*)(hr + kb + 16);
        const float4* bp = (const float4*)(b1 + kb);
        const float4* bq = (const float4*)(b1 + kb + 16);
        float4 f0 = p[0], f1 = p[1], f2 = q[0], f3 = q[1];
        float4 g0 = bp[0], g1 = bp[1], g2 = bq[0], g3 = bq[1];
        float t[16] = { f0.x+g0.x, f0.y+g0.y, f0.z+g0.z, f0.w+g0.w,
                        f1.x+g1.x, f1.y+g1.y, f1.z+g1.z, f1.w+g1.w,
                        f2.x+g2.x, f2.y+g2.y, f2.z+g2.z, f2.w+g2.w,
                        f3.x+g3.x, f3.y+g3.y, f3.z+g3.z, f3.w+g3.w };
        v16bf a2;
#pragma unroll
        for (int j = 0; j < 16; ++j) { float v = t[j]; a2[j] = (__bf16)(v > 0.0f ? v : 0.0f); }
        v16bf bv = *(const v16bf*)(w2p + ((size_t)kk * 32 + lane) * 16);
        c2 = __builtin_amdgcn_wmma_f32_16x16x32_bf16(
            false, a2, false, bv, (short)0, c2, false, false);
    }
    float bias2 = b2[lm];
    int row0 = tile * 16;
#pragma unroll
    for (int r = 0; r < 8; ++r) {
        int m = r + half * 8;
        if (row0 + m < M) {
            float v = c2[r] + bias2;
            embOut[(size_t)(row0 + m) * EMB + lm] = v > 0.0f ? v : 0.0f;
        }
    }
}

__global__ void copy2_kernel(const float* __restrict__ src, float* __restrict__ a,
                             float* __restrict__ b, int n) {
    int i = blockIdx.x * blockDim.x + threadIdx.x;
    if (i < n) { float v = src[i]; a[i] = v; b[i] = v; }
}

__global__ void zero_kernel(float* __restrict__ p, int n) {
    int i = blockIdx.x * blockDim.x + threadIdx.x;
    if (i < n) p[i] = 0.0f;
}

// One lane per (edge, component): e_out[row][c] += val * e_in[col][c]
__global__ __launch_bounds__(256)
void scatter_kernel(const int* __restrict__ rows, const int* __restrict__ cols,
                    const float* __restrict__ vals, const float* __restrict__ ein,
                    float* __restrict__ eout, int nE) {
    long long t = (long long)blockIdx.x * blockDim.x + threadIdx.x;
    int e = (int)(t >> 4);
    if (e >= nE) return;
    int c = (int)(t & 15);
    int rr = rows[e], cc = cols[e];
    float v = vals[e];
    atomicAdd(&eout[(size_t)rr * EMB + c], v * ein[(size_t)cc * EMB + c]);
}

__global__ void accadd_kernel(float* __restrict__ acc, const float* __restrict__ e, int n) {
    int i = blockIdx.x * blockDim.x + threadIdx.x;
    if (i < n) acc[i] += e[i];
}

// Final MLP: pair(B x 32) @ fw1(32x64) -> relu -> @ fw2(64x1) -> relu -> scale+shift
__global__ __launch_bounds__(256)
void final_mlp_kernel(const float* __restrict__ acc4, const int* __restrict__ users,
                      const int* __restrict__ items,
                      const __bf16* __restrict__ fw1p, const float* __restrict__ fb1,
                      const float* __restrict__ fw2, const float* __restrict__ fb2,
                      const float* __restrict__ stds, const float* __restrict__ means,
                      float* __restrict__ out, int B) {
    int lane = threadIdx.x & 31;
    int half = lane >> 4, lm = lane & 15;
    int tile = uni(blockIdx.x * 8 + (int)(threadIdx.x >> 5));
    if (tile * 16 >= B) return;
    int i = tile * 16 + lm;
    int u  = users[i], it = items[i];
    const float* lu = acc4 + (size_t)u * EMB;
    const float* li = acc4 + (size_t)(N_USERS + it) * EMB;

    v16bf a;                                      // pair row, light = acc/4 folded in
#pragma unroll
    for (int j = 0; j < 16; ++j) {
        int k = kmap(j, half);                    // 0..31
        float v = (k < 16 ? lu[k] : li[k - 16]) * 0.25f;
        a[j] = (__bf16)v;
    }
    v8f h[4] = {v8f{}, v8f{}, v8f{}, v8f{}};
#pragma unroll
    for (int nt = 0; nt < 4; ++nt) {
        v16bf b = *(const v16bf*)(fw1p + ((size_t)nt * 32 + lane) * 16);
        h[nt] = __builtin_amdgcn_wmma_f32_16x16x32_bf16(
            false, a, false, b, (short)0, h[nt], false, false);
    }
    // relu(h + fb1) dot fw2 : per-lane partials over this lane's 4 columns
    float p[8] = {0,0,0,0,0,0,0,0};
#pragma unroll
    for (int nt = 0; nt < 4; ++nt) {
        int n = nt * 16 + lm;
        float bias = fb1[n], w = fw2[n];
#pragma unroll
        for (int r = 0; r < 8; ++r) {
            float hv = h[nt][r] + bias;
            p[r] += (hv > 0.0f ? hv : 0.0f) * w;
        }
    }
    // reduce across the 16 lanes sharing the same row set (width-16 xor groups)
#pragma unroll
    for (int m = 1; m < 16; m <<= 1) {
#pragma unroll
        for (int r = 0; r < 8; ++r) p[r] += __shfl_xor(p[r], m, 16);
    }
    if (lm == 0) {
#pragma unroll
        for (int r = 0; r < 8; ++r) {
            int idx = tile * 16 + half * 8 + r;
            float g = p[r] + fb2[0];
            g = g > 0.0f ? g : 0.0f;
            int uu = users[idx];
            out[idx] = g * stds[uu] + means[uu];
        }
    }
}

extern "C" void kernel_launch(void* const* d_in, const int* in_sizes, int n_in,
                              void* d_out, int out_size, void* d_ws, size_t ws_size,
                              hipStream_t stream) {
    const float* user_feats = (const float*)d_in[0];
    const float* movie_feats= (const float*)d_in[1];
    const int*   g_rows     = (const int*)  d_in[2];
    const int*   g_cols     = (const int*)  d_in[3];
    const float* g_vals     = (const float*)d_in[4];
    const float* uw1 = (const float*)d_in[5];  const float* ub1 = (const float*)d_in[6];
    const float* uw2 = (const float*)d_in[7];  const float* ub2 = (const float*)d_in[8];
    const float* mw1 = (const float*)d_in[9];  const float* mb1 = (const float*)d_in[10];
    const float* mw2 = (const float*)d_in[11]; const float* mb2 = (const float*)d_in[12];
    const float* fw1 = (const float*)d_in[13]; const float* fb1 = (const float*)d_in[14];
    const float* fw2 = (const float*)d_in[15]; const float* fb2 = (const float*)d_in[16];
    const float* means = (const float*)d_in[17];
    const float* stds  = (const float*)d_in[18];
    const int* users = (const int*)d_in[19];
    const int* items = (const int*)d_in[20];
    int nE = in_sizes[2];
    int B  = in_sizes[19];

    const int KT1u = (M_ITEMS + 31) / 32;   // 157 (user MLP: K = 5000)
    const int KT1m = (N_USERS + 31) / 32;   // 313 (movie MLP: K = 10000)
    const int MtU  = (N_USERS + 15) / 16;   // 625
    const int MtM  = (M_ITEMS + 15) / 16;   // 313

    // Workspace layout (256B-aligned slabs)
    size_t off = 0;
    auto alloc = [&](size_t bytes) { size_t o = off; off = (off + bytes + 255) & ~(size_t)255; return o; };
    char* ws = (char*)d_ws;
    float*  emb  = (float*)(ws + alloc((size_t)N_NODES * EMB * 4));
    float*  eA   = (float*)(ws + alloc((size_t)N_NODES * EMB * 4));
    float*  eB   = (float*)(ws + alloc((size_t)N_NODES * EMB * 4));
    float*  acc  = (float*)(ws + alloc((size_t)N_NODES * EMB * 4));
    int hRowsU = MtU * 16, hRowsM = MtM * 16;                    // 10000 + 5008
    float*  hU   = (float*)(ws + alloc((size_t)(hRowsU + hRowsM) * HID * 4));
    float*  hM   = hU + (size_t)hRowsU * HID;
    __bf16* uw1p = (__bf16*)(ws + alloc((size_t)KT1u * 4 * 32 * 16 * 2));
    __bf16* mw1p = (__bf16*)(ws + alloc((size_t)KT1m * 4 * 32 * 16 * 2));
    __bf16* uw2p = (__bf16*)(ws + alloc((size_t)2 * 1 * 32 * 16 * 2));
    __bf16* mw2p = (__bf16*)(ws + alloc((size_t)2 * 1 * 32 * 16 * 2));
    __bf16* fw1p = (__bf16*)(ws + alloc((size_t)1 * 4 * 32 * 16 * 2));

    // 1) Pack weights to bf16 B-fragments; zero h accumulators
    {
        int t;
        t = KT1u * 4 * 32; pack_w_kernel<<<(t + 127) / 128, 128, 0, stream>>>(uw1, M_ITEMS, HID, 4, KT1u, uw1p);
        t = KT1m * 4 * 32; pack_w_kernel<<<(t + 127) / 128, 128, 0, stream>>>(mw1, N_USERS, HID, 4, KT1m, mw1p);
        t = 2 * 1 * 32;    pack_w_kernel<<<(t + 127) / 128, 128, 0, stream>>>(uw2, HID, EMB, 1, 2, uw2p);
        t = 2 * 1 * 32;    pack_w_kernel<<<(t + 127) / 128, 128, 0, stream>>>(mw2, HID, EMB, 1, 2, mw2p);
        t = 1 * 4 * 32;    pack_w_kernel<<<(t + 127) / 128, 128, 0, stream>>>(fw1, 2 * EMB, HID, 4, 1, fw1p);
        int hn = (hRowsU + hRowsM) * HID;
        zero_kernel<<<(hn + 255) / 256, 256, 0, stream>>>(hU, hn);
    }

    // 2) Embedding MLPs: K-split layer-1 partial GEMMs, then layer-2
    {
        int wU = MtU * KSPLIT, wM = MtM * KSPLIT;                // 5000 / 2504 waves
        gemm1_partial_kernel<<<(wU + 7) / 8, 256, 0, stream>>>(
            user_feats, N_USERS, M_ITEMS, KT1u, uw1p, hU);
        gemm1_partial_kernel<<<(wM + 7) / 8, 256, 0, stream>>>(
            movie_feats, M_ITEMS, N_USERS, KT1m, mw1p, hM);
        mlp_stage2_kernel<<<(MtU + 7) / 8, 256, 0, stream>>>(hU, N_USERS, ub1, uw2p, ub2, emb);
        mlp_stage2_kernel<<<(MtM + 7) / 8, 256, 0, stream>>>(hM, M_ITEMS, mb1, mw2p, mb2,
                                                             emb + (size_t)N_USERS * EMB);
    }

    // 3) LightGCN propagation: acc = emb + sum_{l=1..3} S^l emb
    {
        int n = N_NODES * EMB;
        int blk = (n + 255) / 256;
        copy2_kernel<<<blk, 256, 0, stream>>>(emb, eA, acc, n);
        float* cur = eA; float* nxt = eB;
        long long st = (long long)nE * 16;
        int sblk = (int)((st + 255) / 256);
        for (int l = 0; l < 3; ++l) {
            zero_kernel<<<blk, 256, 0, stream>>>(nxt, n);
            scatter_kernel<<<sblk, 256, 0, stream>>>(g_rows, g_cols, g_vals, cur, nxt, nE);
            accadd_kernel<<<blk, 256, 0, stream>>>(acc, nxt, n);
            float* t = cur; cur = nxt; nxt = t;
        }
    }

    // 4) Final scoring MLP (K=32 bf16 WMMA) + denorm epilogue
    {
        int Mt = (B + 15) / 16;          // 6250
        final_mlp_kernel<<<(Mt + 7) / 8, 256, 0, stream>>>(
            acc, users, items, fw1p, fb1, fw2, fb2, stds, means, (float*)d_out, B);
    }
}
